// MPNNModel_41953240547661
// MI455X (gfx1250) — matrix operand order repaired and checked by reading, not compiled, not run on hardware
//
#include <hip/hip_runtime.h>
#include <hip/hip_bf16.h>
#include <math.h>

// ---------------------------------------------------------------------------
// Equivariant MPNN for MI455X (gfx1250, wave32).
// Dense GEMMs -> v_wmma_f32_16x16x32_f16 (f32 accumulate), fully branchless:
// all K dims padded to multiples of 32 (edge features padded 130 -> 160).
// CG tensor product -> VALU with constant-memory Wigner-3j tables.
// Scatter-mean -> global f32 atomics. Weights are L2-resident (192MB L2).
// ---------------------------------------------------------------------------

typedef __attribute__((ext_vector_type(16))) _Float16 v16h;
typedef __attribute__((ext_vector_type(8)))  float    v8f;

#define HDIM 192   // padded node feature width (seq3 = 192)
#define EPAD 160   // edge-feature width padded to multiple of 32 (130 -> 160)

// ------------------------- Wigner 3j constant tables -----------------------
enum { CG000, CG011, CG101, CG110, CG022, CG202, CG220,
       CG111, CG112, CG121, CG211, CG212, CG122, CG221, CG222 };

struct CgT { float c111[27]; float c112[45]; float c212[75]; float c222[125]; };

constexpr CgT build_cg() {
  CgT t{};
  // (1,1,1): Levi-Civita / sqrt(6)
  const float s6 = 0.40824829f;
  t.c111[5]  =  s6; t.c111[15] =  s6; t.c111[19] =  s6;
  t.c111[7]  = -s6; t.c111[11] = -s6; t.c111[21] = -s6;
  // (1,1,2): symmetric-traceless basis S_k / sqrt(5). idx = i*15+j*5+k
  const float a = 0.31622777f, p2 = 0.36514837f, mm = -0.18257419f;
  t.c112[10] = a;  t.c112[30] = a;                       // k=0 (xy)
  t.c112[6]  = a;  t.c112[16] = a;                       // k=1 (yz)
  t.c112[2]  = mm; t.c112[22] = p2; t.c112[42] = mm;     // k=2 (3z^2-1)
  t.c112[28] = a;  t.c112[38] = a;                       // k=3 (xz)
  t.c112[4]  = -a; t.c112[44] = a;                       // k=4 (x^2-y^2)
  // (2,1,2): SO(3) generators on real l=2 basis / sqrt(30). idx=k*15+a*5+k'
  const float s = 0.18257419f, s3 = 0.31622777f, s2 = 0.36514837f;
  t.c212[1]  =  s;  t.c212[15] = -s;  t.c212[49] = -s;   // a=0 (Jy)
  t.c212[47] =  s3; t.c212[63] =  s;  t.c212[33] = -s3;
  t.c212[9]  =  s2; t.c212[65] = -s2; t.c212[23] =  s;   // a=1 (Jz)
  t.c212[51] = -s;
  t.c212[13] = -s;  t.c212[55] =  s;  t.c212[29] = -s;   // a=2 (Jx)
  t.c212[27] = -s3; t.c212[71] =  s;  t.c212[41] =  s3;
  // (2,2,2): symmetric d-symbols, unit Frobenius norm. idx=i*25+j*5+k
  const float v1 = 0.2390457f, v2 = 0.1195229f, v3 = 0.2070196f;
  t.c222[62]  =  v1;
  t.c222[2]   = -v1; t.c222[10]  = -v1; t.c222[50]  = -v1;
  t.c222[74]  = -v1; t.c222[114] = -v1; t.c222[122] = -v1;
  t.c222[32]  =  v2; t.c222[36]  =  v2; t.c222[56]  =  v2;
  t.c222[92]  =  v2; t.c222[88]  =  v2; t.c222[68]  =  v2;
  t.c222[34]  = -v3; t.c222[46]  = -v3; t.c222[106] = -v3;
  t.c222[94]  =  v3; t.c222[98]  =  v3; t.c222[118] =  v3;
  t.c222[8]   =  v3; t.c222[16]  =  v3; t.c222[28]  =  v3;
  t.c222[40]  =  v3; t.c222[76]  =  v3; t.c222[80]  =  v3;
  return t;
}
__constant__ CgT CG = build_cg();

__device__ __forceinline__ float cg_val(int id, int i, int j, int k) {
  const float R3 = 0.57735027f, R5 = 0.44721360f;
  switch (id) {
    case CG000: return 1.f;
    case CG011: return (j == k) ? R3 : 0.f;
    case CG101: return (i == k) ? R3 : 0.f;
    case CG110: return (i == j) ? R3 : 0.f;
    case CG022: return (j == k) ? R5 : 0.f;
    case CG202: return (i == k) ? R5 : 0.f;
    case CG220: return (i == j) ? R5 : 0.f;
    case CG111: return CG.c111[(i*3+j)*3+k];
    case CG112: return CG.c112[(i*3+j)*5+k];
    case CG121: return CG.c112[(i*3+k)*5+j];
    case CG211: return CG.c112[(j*3+k)*5+i];
    case CG212: return CG.c212[(i*3+j)*5+k];
    case CG122: return CG.c212[(j*3+i)*5+k];
    case CG221: return CG.c212[(i*3+k)*5+j];
    case CG222: return CG.c222[(i*5+j)*5+k];
  }
  return 0.f;
}

// ----------------------------- TP path tables ------------------------------
struct TpPath { int cg, o1, m1, d1, oo, mo, dd, l2, woff; float c; };

__constant__ TpPath PATHS[75] = {
  // ---- layer 0 (3 paths), c = sqrt((2l+1)/32)
  {CG000, 0,32,1,  0,32,1, 0,    0, 0.1767767f},
  {CG011, 0,32,1, 32, 8,3, 1, 1024, 0.3061862f},
  {CG022, 0,32,1, 56, 8,5, 2, 1280, 0.3952847f},
  // ---- layer 1 (15 paths)
  {CG000,  0,32,1,   0,32,1, 0,    0, 0.1443376f},
  {CG011,  0,32,1,  32, 8,3, 1, 1024, 0.2314550f},
  {CG022,  0,32,1,  56, 8,5, 2, 1280, 0.2988072f},
  {CG101, 32, 8,3,  32, 8,3, 0, 1536, 0.2314550f},
  {CG110, 32, 8,3,   0,32,1, 1, 1600, 0.1443376f},
  {CG112, 32, 8,3,  56, 8,5, 1, 1856, 0.2988072f},
  {CG111, 32, 8,3,  96, 8,3, 1, 1920, 0.4330127f},
  {CG121, 32, 8,3,  32, 8,3, 2, 1984, 0.2314550f},
  {CG122, 32, 8,3, 120, 8,5, 2, 2048, 0.5590170f},
  {CG202, 56, 8,5,  56, 8,5, 0, 2112, 0.2988072f},
  {CG211, 56, 8,5,  32, 8,3, 1, 2176, 0.2314550f},
  {CG212, 56, 8,5, 120, 8,5, 1, 2240, 0.5590170f},
  {CG220, 56, 8,5,   0,32,1, 2, 2304, 0.1443376f},
  {CG222, 56, 8,5,  56, 8,5, 2, 2560, 0.2988072f},
  {CG221, 56, 8,5,  96, 8,3, 2, 2624, 0.4330127f},
  // ---- layer 2 (27 paths)
  {CG000,   0,32,1,   0,32,1, 0,    0, 0.1443376f},
  {CG011,   0,32,1,  32, 8,3, 1, 1024, 0.2041241f},
  {CG022,   0,32,1,  56, 8,5, 2, 1280, 0.2635231f},
  {CG101,  32, 8,3,  32, 8,3, 0, 1536, 0.2041241f},
  {CG110,  32, 8,3,   0,32,1, 1, 1600, 0.1443376f},
  {CG112,  32, 8,3,  56, 8,5, 1, 1856, 0.2635231f},
  {CG111,  32, 8,3,  96, 8,3, 1, 1920, 0.2738613f},
  {CG121,  32, 8,3,  32, 8,3, 2, 1984, 0.2041241f},
  {CG122,  32, 8,3, 120, 8,5, 2, 2048, 0.3535534f},
  {CG202,  56, 8,5,  56, 8,5, 0, 2112, 0.2635231f},
  {CG211,  56, 8,5,  32, 8,3, 1, 2176, 0.2041241f},
  {CG212,  56, 8,5, 120, 8,5, 1, 2240, 0.3535534f},
  {CG220,  56, 8,5,   0,32,1, 2, 2304, 0.1443376f},
  {CG222,  56, 8,5,  56, 8,5, 2, 2560, 0.2635231f},
  {CG221,  56, 8,5,  96, 8,3, 2, 2624, 0.2738613f},
  {CG101,  96, 8,3,  96, 8,3, 0, 2688, 0.2738613f},
  {CG111,  96, 8,3,  32, 8,3, 1, 2752, 0.2041241f},
  {CG112,  96, 8,3, 120, 8,5, 1, 2816, 0.3535534f},
  {CG110,  96, 8,3, 160,32,1, 1, 2880, 0.2500000f},
  {CG122,  96, 8,3,  56, 8,5, 2, 3136, 0.2635231f},
  {CG121,  96, 8,3,  96, 8,3, 2, 3200, 0.2738613f},
  {CG202, 120, 8,5, 120, 8,5, 0, 3264, 0.3535534f},
  {CG212, 120, 8,5,  56, 8,5, 1, 3328, 0.2635231f},
  {CG211, 120, 8,5,  96, 8,3, 1, 3392, 0.2738613f},
  {CG221, 120, 8,5,  32, 8,3, 2, 3456, 0.2041241f},
  {CG222, 120, 8,5, 120, 8,5, 2, 3520, 0.3535534f},
  {CG220, 120, 8,5, 160,32,1, 2, 3584, 0.2500000f},
  // ---- layer 3 (30 paths, fans differ for io3/io4/io5)
  {CG000,   0,32,1,   0,32,1, 0,    0, 0.1443376f},
  {CG011,   0,32,1,  32, 8,3, 1, 1024, 0.2041241f},
  {CG022,   0,32,1,  56, 8,5, 2, 1280, 0.2635231f},
  {CG101,  32, 8,3,  32, 8,3, 0, 1536, 0.2041241f},
  {CG110,  32, 8,3,   0,32,1, 1, 1600, 0.1443376f},
  {CG112,  32, 8,3,  56, 8,5, 1, 1856, 0.2635231f},
  {CG111,  32, 8,3,  96, 8,3, 1, 1920, 0.2041241f},
  {CG121,  32, 8,3,  32, 8,3, 2, 1984, 0.2041241f},
  {CG122,  32, 8,3, 120, 8,5, 2, 2048, 0.2635231f},
  {CG202,  56, 8,5,  56, 8,5, 0, 2112, 0.2635231f},
  {CG211,  56, 8,5,  32, 8,3, 1, 2176, 0.2041241f},
  {CG212,  56, 8,5, 120, 8,5, 1, 2240, 0.2635231f},
  {CG220,  56, 8,5,   0,32,1, 2, 2304, 0.1443376f},
  {CG222,  56, 8,5,  56, 8,5, 2, 2560, 0.2635231f},
  {CG221,  56, 8,5,  96, 8,3, 2, 2624, 0.2041241f},
  {CG101,  96, 8,3,  96, 8,3, 0, 2688, 0.2041241f},
  {CG111,  96, 8,3,  32, 8,3, 1, 2752, 0.2041241f},
  {CG112,  96, 8,3, 120, 8,5, 1, 2816, 0.2635231f},
  {CG110,  96, 8,3, 160,32,1, 1, 2880, 0.1443376f},
  {CG122,  96, 8,3,  56, 8,5, 2, 3136, 0.2635231f},
  {CG121,  96, 8,3,  96, 8,3, 2, 3200, 0.2041241f},
  {CG202, 120, 8,5, 120, 8,5, 0, 3264, 0.2635231f},
  {CG212, 120, 8,5,  56, 8,5, 1, 3328, 0.2635231f},
  {CG211, 120, 8,5,  96, 8,3, 1, 3392, 0.2041241f},
  {CG221, 120, 8,5,  32, 8,3, 2, 3456, 0.2041241f},
  {CG222, 120, 8,5, 120, 8,5, 2, 3520, 0.2635231f},
  {CG220, 120, 8,5, 160,32,1, 2, 3584, 0.1443376f},
  {CG000, 160,32,1, 160,32,1, 0, 3840, 0.1443376f},
  {CG011, 160,32,1,  96, 8,3, 1, 4864, 0.2041241f},
  {CG022, 160,32,1, 120, 8,5, 2, 5120, 0.2635231f},
};

// ---------------------- per-layer irrep (batch-norm) info ------------------
struct IrInfo { int off, m, d, center, chbase; };
__constant__ IrInfo IRS[4][6] = {
  {{0,32,1,1,0},{32,8,3,0,32},{56,8,5,0,40},{0,0,1,0,0},{0,0,1,0,0},{0,0,1,0,0}},
  {{0,32,1,1,0},{32,8,3,0,32},{56,8,5,0,40},{96,8,3,0,48},{120,8,5,0,56},{0,0,1,0,0}},
  {{0,32,1,1,0},{32,8,3,0,32},{56,8,5,0,40},{96,8,3,0,48},{120,8,5,0,56},{160,32,1,0,64}},
  {{0,32,1,1,0},{32,8,3,0,32},{56,8,5,0,40},{96,8,3,0,48},{120,8,5,0,56},{160,32,1,0,64}},
};
__constant__ int NIRS[4] = {3, 5, 6, 6};

// ------------------------------- WMMA GEMM ---------------------------------
// One wave (32 threads) per 16x16 output tile. C = act(A(MxK) @ W(KxN) + b).
// K is always a multiple of 32 (edge features padded to 160), so the loop is
// fully branchless: float2 A loads, unmasked strided B loads, one wmma/chunk.
// act: 0 = none, 1 = relu, 2 = tanh.
__global__ __launch_bounds__(32)
void gemm_wmma(const float* __restrict__ A, int lda,
               const float* __restrict__ W, int ldw,
               const float* __restrict__ bias,
               float* __restrict__ C, int ldc,
               int K, int act) {
  const int tile_m = blockIdx.x * 16;
  const int tile_n = blockIdx.y * 16;
  const int lane = threadIdx.x;      // 0..31 (wave32)
  const int half = lane >> 4;        // 0 / 1
  const int mr   = lane & 15;

  v8f acc = {};
  const float* arow = A + (size_t)(tile_m + mr) * lda;
  const int col = tile_n + mr;

  // A 16x32 per-lane layout (ISA 7.12.2): lanes<16 hold K 0..7,16..23;
  // lanes>=16 hold K 8..15,24..31. Pairs (kl,kl+1) are contiguous -> float2.
  for (int k0 = 0; k0 < K; k0 += 32) {
    v16h va, vb;
    #pragma unroll
    for (int v = 0; v < 8; ++v) {
      const int kl = (v < 4) ? (half * 8 + v * 2) : (16 + half * 8 + (v - 4) * 2);
      const float2 f2 = *(const float2*)(arow + k0 + kl);
      va[2*v]   = (_Float16)f2.x;
      va[2*v+1] = (_Float16)f2.y;
    }
    // B 32x16: same K order as A; lanes<16 hold K 0..7,16..23; lanes>=16 hold K 8..15,24..31
    #pragma unroll
    for (int v = 0; v < 8; ++v) {
      const int k = k0 + ((v < 4) ? (half * 8 + 2 * v) : (16 + half * 8 + 2 * (v - 4)));
      vb[2*v]   = (_Float16)W[(size_t)k       * ldw + col];
      vb[2*v+1] = (_Float16)W[(size_t)(k + 1) * ldw + col];
    }
    acc = __builtin_amdgcn_wmma_f32_16x16x32_f16(false, va, false, vb,
                                                 (short)0, acc, false, false);
  }
  const float bv = bias ? bias[col] : 0.f;
  #pragma unroll
  for (int r = 0; r < 8; ++r) {
    const int row = tile_m + r + half * 8;  // C: VGPR r -> M=r / M=8+r
    float v = acc[r] + bv;
    if (act == 1) v = v > 0.f ? v : 0.f;
    else if (act == 2) v = tanhf(v);
    C[(size_t)row * ldc + col] = v;
  }
}

// ------------------------------ misc kernels -------------------------------
__global__ void fill_zero(float* p, long long n) {
  long long i = (long long)blockIdx.x * blockDim.x + threadIdx.x;
  if (i < n) p[i] = 0.f;
}

__global__ void edge_geom(const float* __restrict__ pos,
                          const float* __restrict__ eattr,
                          const int* __restrict__ src, const int* __restrict__ dst,
                          float* __restrict__ eain, float* __restrict__ shb, int E) {
  int e = blockIdx.x * blockDim.x + threadIdx.x;
  if (e >= E) return;
  int s = src[e], d = dst[e];
  float vx = pos[d*3+0] - pos[s*3+0];
  float vy = pos[d*3+1] - pos[s*3+1];
  float vz = pos[d*3+2] - pos[s*3+2];
  float dist = sqrtf(vx*vx + vy*vy + vz*vz + 1e-12f);
  for (int i = 0; i < 80; ++i) eain[(size_t)e*EPAD + i] = eattr[(size_t)e*80 + i];
  const float step = 5.0f / 49.0f;
  const float coeff = -0.5f / (step * step);
  for (int j = 0; j < 50; ++j) {
    float dv = dist - step * j;
    eain[(size_t)e*EPAD + 80 + j] = expf(coeff * dv * dv);
  }
  for (int j = 130; j < EPAD; ++j) eain[(size_t)e*EPAD + j] = 0.f;  // K padding
  float ux = vx / dist, uy = vy / dist, uz = vz / dist;
  const float s3 = 1.7320508f, s15 = 3.8729833f, s5 = 2.2360680f;
  float* sh = shb + (size_t)e * 9;
  sh[0] = 1.f;        sh[1] = s3 * uy;     sh[2] = s3 * uz;  sh[3] = s3 * ux;
  sh[4] = s15*ux*uy;  sh[5] = s15*uy*uz;   sh[6] = 0.5f*s5*(3.f*uz*uz - 1.f);
  sh[7] = s15*ux*uz;  sh[8] = 0.5f*s15*(ux*ux - uy*uy);
}

__global__ void gather_ein(const float* __restrict__ ea, const float* __restrict__ h,
                           const int* __restrict__ src, const int* __restrict__ dst,
                           float* __restrict__ ein, int E) {
  int e = blockIdx.x * blockDim.x + threadIdx.x;
  if (e >= E) return;
  int s = src[e], d = dst[e];
  for (int i = 0; i < 32; ++i) {
    ein[(size_t)e*96 + i]      = ea[(size_t)e*32 + i];
    ein[(size_t)e*96 + 32 + i] = h[(size_t)s*HDIM + i];
    ein[(size_t)e*96 + 64 + i] = h[(size_t)d*HDIM + i];
  }
}

// Fused: weight generation (hidden96 @ fcW2 + b2) + CG tensor product +
// scatter (atomic add by src). One thread per edge.
__global__ void tp_scatter(const float* __restrict__ hid,   // E x 96 (post-ReLU)
                           const float* __restrict__ hprev, // N x 192
                           const float* __restrict__ shb,   // E x 9
                           const int* __restrict__ src, const int* __restrict__ dst,
                           const float* __restrict__ W2, const float* __restrict__ b2,
                           int nw, float* __restrict__ acc, float* __restrict__ cnt,
                           int pbase, int pcount, int E, int outdim) {
  int e = blockIdx.x * blockDim.x + threadIdx.x;
  if (e >= E) return;
  float hv[96];
  for (int i = 0; i < 96; ++i) hv[i] = hid[(size_t)e*96 + i];
  float shv[9];
  for (int i = 0; i < 9; ++i) shv[i] = shb[(size_t)e*9 + i];
  const int s = src[e], d = dst[e];
  const float* x = hprev + (size_t)d * HDIM;
  float msg[HDIM];
  for (int i = 0; i < HDIM; ++i) msg[i] = 0.f;

  for (int p = 0; p < pcount; ++p) {
    const TpPath P = PATHS[pbase + p];
    const int d2 = 2 * P.l2 + 1;
    const float* shl = shv + P.l2 * P.l2;
    float t[32][5];
    for (int u = 0; u < P.m1; ++u) {
      const float* xu = x + P.o1 + u * P.d1;
      for (int ko = 0; ko < P.dd; ++ko) {
        float sacc = 0.f;
        for (int i = 0; i < P.d1; ++i)
          for (int j = 0; j < d2; ++j) {
            float cgv = cg_val(P.cg, i, j, ko);
            if (cgv != 0.f) sacc += cgv * xu[i] * shl[j];
          }
        t[u][ko] = sacc;
      }
    }
    for (int w = 0; w < P.mo; ++w) {
      float* out = msg + P.oo + w * P.dd;
      for (int u = 0; u < P.m1; ++u) {
        int colw = P.woff + u * P.mo + w;
        __builtin_prefetch(&W2[(size_t)48 * nw + colw], 0, 1); // L2-resident weights
        float wv = b2[colw];
        for (int hh = 0; hh < 96; ++hh) wv += hv[hh] * W2[(size_t)hh * nw + colw];
        const float cw = P.c * wv;
        for (int ko = 0; ko < P.dd; ++ko) out[ko] += cw * t[u][ko];
      }
    }
  }
  float* arow = acc + (size_t)s * HDIM;
  for (int i = 0; i < outdim; ++i) atomicAdd(&arow[i], msg[i]);
  atomicAdd(&cnt[s], 1.f);
}

__global__ void finalize_scatter(float* __restrict__ hnext, const float* __restrict__ hprev,
                                 const float* __restrict__ cnt, int N) {
  long long i = (long long)blockIdx.x * blockDim.x + threadIdx.x;
  if (i >= (long long)N * HDIM) return;
  int n = (int)(i / HDIM);
  float c = cnt[n]; c = c < 1.f ? 1.f : c;
  hnext[i] = hnext[i] / c + hprev[i];
}

__global__ void bn_mean(const float* __restrict__ h, float* __restrict__ mean, int N) {
  __shared__ float sm[256];
  int c = blockIdx.x;                       // scalar-even channels: cols 0..31
  float s = 0.f;
  for (int n = threadIdx.x; n < N; n += blockDim.x) s += h[(size_t)n*HDIM + c];
  sm[threadIdx.x] = s; __syncthreads();
  for (int st = 128; st > 0; st >>= 1) {
    if ((int)threadIdx.x < st) sm[threadIdx.x] += sm[threadIdx.x + st];
    __syncthreads();
  }
  if (threadIdx.x == 0) mean[c] = sm[0] / (float)N;
}

__global__ void bn_norm(const float* __restrict__ h, const float* __restrict__ mean,
                        float* __restrict__ inv, int N, int layer) {
  __shared__ float sm[256];
  int ch = blockIdx.x;
  IrInfo I = IRS[layer][0];
  for (int t = 0; t < NIRS[layer]; ++t) {
    IrInfo J = IRS[layer][t];
    if (ch >= J.chbase && ch < J.chbase + J.m) { I = J; break; }
  }
  int u = ch - I.chbase;
  float mu = I.center ? mean[ch] : 0.f;
  float s = 0.f;
  for (int n = threadIdx.x; n < N; n += blockDim.x) {
    const float* f = h + (size_t)n*HDIM + I.off + u * I.d;
    for (int k = 0; k < I.d; ++k) { float v = f[k] - mu; s += v * v; }
  }
  sm[threadIdx.x] = s; __syncthreads();
  for (int st = 128; st > 0; st >>= 1) {
    if ((int)threadIdx.x < st) sm[threadIdx.x] += sm[threadIdx.x + st];
    __syncthreads();
  }
  if (threadIdx.x == 0) inv[ch] = rsqrtf(sm[0] / (float)(N * I.d) + 1e-5f);
}

__global__ void bn_apply(float* __restrict__ h, const float* __restrict__ mean,
                         const float* __restrict__ inv, int N, int layer, int outdim) {
  long long i = (long long)blockIdx.x * blockDim.x + threadIdx.x;
  if (i >= (long long)N * outdim) return;
  int n = (int)(i / outdim), c = (int)(i % outdim);
  int ch = 0, center = 0;
  for (int t = 0; t < NIRS[layer]; ++t) {
    IrInfo J = IRS[layer][t];
    if (c >= J.off && c < J.off + J.m * J.d) {
      ch = J.chbase + (c - J.off) / J.d; center = J.center; break;
    }
  }
  float v = h[(size_t)n*HDIM + c];
  if (center) v -= mean[ch];
  h[(size_t)n*HDIM + c] = v * inv[ch];
}

// ------------------------------ host launcher ------------------------------
static inline void launch_gemm(const float* A, int lda, const float* W, int ldw,
                               const float* b, float* C, int ldc,
                               int M, int Ncols, int K, int act, hipStream_t st) {
  dim3 grid(M / 16, Ncols / 16, 1);
  gemm_wmma<<<grid, 32, 0, st>>>(A, lda, W, ldw, b, C, ldc, K, act);
}

extern "C" void kernel_launch(void* const* d_in, const int* in_sizes, int n_in,
                              void* d_out, int out_size, void* d_ws, size_t ws_size,
                              hipStream_t stream) {
  const float* x     = (const float*)d_in[0];
  const float* pos   = (const float*)d_in[1];
  const float* eattr = (const float*)d_in[2];
  const float* nW1 = (const float*)d_in[3];  const float* nb1 = (const float*)d_in[4];
  const float* nW2 = (const float*)d_in[5];  const float* nb2 = (const float*)d_in[6];
  const float* eW1 = (const float*)d_in[7];  const float* eb1 = (const float*)d_in[8];
  const float* eW2 = (const float*)d_in[9];  const float* eb2 = (const float*)d_in[10];
  const float *fcW1[4], *fcb1[4], *fcW2[4], *fcb2[4];
  for (int i = 0; i < 4; ++i) {
    fcW1[i] = (const float*)d_in[11 + 4*i]; fcb1[i] = (const float*)d_in[12 + 4*i];
    fcW2[i] = (const float*)d_in[13 + 4*i]; fcb2[i] = (const float*)d_in[14 + 4*i];
  }
  const float* fW1 = (const float*)d_in[27];
  const float* fW2 = (const float*)d_in[28];
  const int* eidx  = (const int*)d_in[29];

  const int N = in_sizes[0] / 64;     // 10000
  const int E = in_sizes[2] / 80;     // 20000
  const int* src = eidx;
  const int* dst = eidx + E;

  // workspace layout (floats)
  float* w0    = (float*)d_ws;
  float* hA    = w0;                          // N*192
  float* hB    = hA  + (size_t)N * HDIM;      // N*192
  float* ea    = hB  + (size_t)N * HDIM;      // E*32
  float* shb   = ea  + (size_t)E * 32;        // E*9
  float* big   = shb + (size_t)E * 9;         // E*160 (edge-MLP input / e_in)
  float* scr   = big + (size_t)E * EPAD;      // E*96  (hidden / general scratch)
  float* cnt   = scr + (size_t)E * 96;        // N
  float* stats = cnt + N;                     // 32 mean + 96 inv
  float* mean  = stats;
  float* inv   = stats + 32;

  const int nwv[4]   = {1536, 2688, 3840, 5376};
  const int outd[4]  = {96, 160, 192, 192};
  const int pbase[4] = {0, 3, 18, 45};
  const int pcnt[4]  = {3, 15, 27, 30};
  const int nch[4]   = {48, 64, 96, 96};

  const long long hlen = (long long)N * HDIM;
  fill_zero<<<(int)((hlen + 255) / 256), 256, 0, stream>>>(hA, hlen);

  // node embedding: relu(x @ nW1 + nb1) @ nW2 + nb2 -> hA[:, :32]
  launch_gemm(x, 64, nW1, 32, nb1, scr, 32, N, 32, 64, 1, stream);
  launch_gemm(scr, 32, nW2, 32, nb2, hA, HDIM, N, 32, 32, 0, stream);

  // edge geometry + gaussian basis + spherical harmonics (padded to 160)
  edge_geom<<<(E + 255) / 256, 256, 0, stream>>>(pos, eattr, src, dst, big, shb, E);

  // edge embedding: relu(eain @ eW1 + eb1) @ eW2 + eb2 -> ea   (K=160 padded)
  launch_gemm(big, EPAD, eW1, 32, eb1, scr, 32, E, 32, EPAD, 1, stream);
  launch_gemm(scr, 32, eW2, 32, eb2, ea, 32, E, 32, 32, 0, stream);

  float* hprev = hA;
  float* hnext = hB;
  for (int li = 0; li < 4; ++li) {
    // e_in = [ea | h[src,:32] | h[dst,:32]]
    gather_ein<<<(E + 255) / 256, 256, 0, stream>>>(ea, hprev, src, dst, big, E);
    // fc hidden: relu(e_in @ W1 + b1) -> scr (E x 96)
    launch_gemm(big, 96, fcW1[li], 96, fcb1[li], scr, 96, E, 96, 96, 1, stream);
    // zero accumulators
    fill_zero<<<(int)((hlen + 255) / 256), 256, 0, stream>>>(hnext, hlen);
    fill_zero<<<(N + 255) / 256, 256, 0, stream>>>(cnt, N);
    // fused weight-gen + tensor product + scatter
    tp_scatter<<<(E + 127) / 128, 128, 0, stream>>>(scr, hprev, shb, src, dst,
                                                    fcW2[li], fcb2[li], nwv[li],
                                                    hnext, cnt, pbase[li], pcnt[li],
                                                    E, outd[li]);
    // mean + residual
    finalize_scatter<<<(int)((hlen + 255) / 256), 256, 0, stream>>>(hnext, hprev, cnt, N);
    // equivariant batch norm
    bn_mean<<<32, 256, 0, stream>>>(hnext, mean, N);
    bn_norm<<<nch[li], 256, 0, stream>>>(hnext, mean, inv, N, li);
    long long alen = (long long)N * outd[li];
    bn_apply<<<(int)((alen + 255) / 256), 256, 0, stream>>>(hnext, mean, inv, N, li, outd[li]);
    // swap
    float* tmp = hprev; hprev = hnext; hnext = tmp;
  }

  // head: tanh(h @ fW1) @ fW2 -> d_out
  launch_gemm(hprev, HDIM, fW1, 64, nullptr, scr, 64, N, 64, HDIM, 2, stream);
  launch_gemm(scr, 64, fW2, 64, nullptr, (float*)d_out, 64, N, 64, 64, 0, stream);

  (void)n_in; (void)out_size; (void)ws_size;
}
